// RelativeMultiHeadAttention_27444841022164
// MI455X (gfx1250) — compile-verified, matrix operand order checked by reading, not run on hardware
//
#include <hip/hip_runtime.h>
#include <cstdint>
#include <cstddef>

// ---- problem constants ----
#define D_MODEL 1024
#define D_HEAD  64
#define N_HEADS 16
#define BATCH   4
#define TLEN    1024
#define NEGINF  (-1e9f)

typedef unsigned short u16;
typedef __attribute__((ext_vector_type(16))) __bf16 v16bf;
typedef __attribute__((ext_vector_type(8)))  float  v8f;
typedef __attribute__((ext_vector_type(8)))  u16    u16x8;
typedef __attribute__((ext_vector_type(16))) u16    u16x16;

__device__ __forceinline__ u16 f2bf(float f) {
  unsigned int u = __float_as_uint(f);
  u += 0x7FFFu + ((u >> 16) & 1u);          // round-to-nearest-even
  return (u16)(u >> 16);
}

__device__ __forceinline__ u16x16 cat8(u16x8 a, u16x8 b) {
  u16x16 r;
#pragma unroll
  for (int i = 0; i < 8; ++i) { r[i] = a[i]; r[i + 8] = b[i]; }
  return r;
}

// D = A(16x32 bf16) * B(32x16 bf16) + C (f32)
__device__ __forceinline__ v8f wmma_bf16(u16x16 a, u16x16 b, v8f c) {
  return __builtin_amdgcn_wmma_f32_16x16x32_bf16(
      false, __builtin_bit_cast(v16bf, a),
      false, __builtin_bit_cast(v16bf, b),
      (short)0, c, false, false);
}

// ------------------------------------------------------------------
// 1) LayerNorm: one block per row of 1024
// ------------------------------------------------------------------
__global__ __launch_bounds__(256) void ln_kernel(const float* __restrict__ x,
                                                 const float* __restrict__ gamma,
                                                 const float* __restrict__ beta,
                                                 float* __restrict__ xn_f,
                                                 u16*   __restrict__ xn_b) {
  const int row = blockIdx.x;
  const int tid = threadIdx.x;
  const float* xr = x + (size_t)row * D_MODEL;
  __shared__ float red[256];

  float v[4];
  float s = 0.f;
#pragma unroll
  for (int i = 0; i < 4; ++i) { v[i] = xr[tid + 256 * i]; s += v[i]; }
  red[tid] = s; __syncthreads();
  for (int off = 128; off > 0; off >>= 1) {
    if (tid < off) red[tid] += red[tid + off];
    __syncthreads();
  }
  const float mean = red[0] * (1.f / D_MODEL);
  __syncthreads();

  float vs = 0.f;
#pragma unroll
  for (int i = 0; i < 4; ++i) { float d = v[i] - mean; vs += d * d; }
  red[tid] = vs; __syncthreads();
  for (int off = 128; off > 0; off >>= 1) {
    if (tid < off) red[tid] += red[tid + off];
    __syncthreads();
  }
  const float inv = rsqrtf(red[0] * (1.f / D_MODEL) + 1e-5f);

#pragma unroll
  for (int i = 0; i < 4; ++i) {
    const int c = tid + 256 * i;
    const float y = (v[i] - mean) * inv * gamma[c] + beta[c];
    xn_f[(size_t)row * D_MODEL + c] = y;
    xn_b[(size_t)row * D_MODEL + c] = f2bf(y);
  }
}

// ------------------------------------------------------------------
// 2) f32 -> bf16 convert
// ------------------------------------------------------------------
__global__ __launch_bounds__(256) void cvt_kernel(const float* __restrict__ src,
                                                  u16* __restrict__ dst, int n) {
  int i = blockIdx.x * 256 + threadIdx.x;
  if (i < n) dst[i] = f2bf(src[i]);
}

// ------------------------------------------------------------------
// 3) GEMM: C[M,N] = A[M,K] * W[N,K]^T, M=4096, N=K=1024, bf16 in, f32 acc
//    mode 0: bf16 out ; mode 1: f32 out + bias[n] + resid[m,n]
//    block tile 128x128, 8 waves, each wave 64x32 (4x2 wmma tiles).
//    LDS double-buffered: stage chunk i+1 while WMMAs consume chunk i.
// ------------------------------------------------------------------
struct __align__(16) GemmSmem { u16 As[128 * 40]; u16 Bs[128 * 40]; };

__device__ __forceinline__ void gemm_stage(GemmSmem* buf,
                                           const u16* __restrict__ A,
                                           const u16* __restrict__ W,
                                           size_t mBlk, size_t nBlk,
                                           int k0, int tid) {
#pragma unroll
  for (int c = 0; c < 2; ++c) {
    const int ch = tid * 2 + c;          // 0..511
    const int r  = ch >> 2;
    const int kc = ch & 3;
    *(u16x8*)&buf->As[r * 40 + kc * 8] =
        *(const u16x8*)&A[(mBlk + r) * D_MODEL + k0 + kc * 8];
    *(u16x8*)&buf->Bs[r * 40 + kc * 8] =
        *(const u16x8*)&W[(nBlk + r) * D_MODEL + k0 + kc * 8];
  }
}

__global__ __launch_bounds__(256) void gemm_bf16(const u16* __restrict__ A,
                                                 const u16* __restrict__ W,
                                                 u16*   __restrict__ Cb,
                                                 float* __restrict__ Cf,
                                                 const float* __restrict__ bias,
                                                 const float* __restrict__ resid,
                                                 int mode) {
  __shared__ GemmSmem sm[2];
  const int tid  = threadIdx.x;
  const int lane = tid & 31;
  const int wave = tid >> 5;
  const int waveM = wave >> 2;   // 0..1  -> 64 rows each
  const int waveN = wave & 3;    // 0..3  -> 32 cols each
  const int half  = lane >> 4;
  const int l15   = lane & 15;

  const size_t mBlk = (size_t)blockIdx.y * 128;
  const size_t nBlk = (size_t)blockIdx.x * 128;

  v8f acc[4][2] = {};

  gemm_stage(&sm[0], A, W, mBlk, nBlk, 0, tid);
  __syncthreads();

  int cur = 0;
  for (int k0 = 0; k0 < D_MODEL; k0 += 32) {
    if (k0 + 32 < D_MODEL)
      gemm_stage(&sm[cur ^ 1], A, W, mBlk, nBlk, k0 + 32, tid);

    const GemmSmem* rd = &sm[cur];
    u16x16 af[4], bf[2];
#pragma unroll
    for (int mt = 0; mt < 4; ++mt) {
      const int row = waveM * 64 + mt * 16 + l15;
      af[mt] = cat8(*(const u16x8*)&rd->As[row * 40 + half * 8],
                    *(const u16x8*)&rd->As[row * 40 + 16 + half * 8]);
    }
#pragma unroll
    for (int nt = 0; nt < 2; ++nt) {
      const int col = waveN * 32 + nt * 16 + l15;
      bf[nt] = cat8(*(const u16x8*)&rd->Bs[col * 40 + half * 8],
                    *(const u16x8*)&rd->Bs[col * 40 + 16 + half * 8]);
    }
#pragma unroll
    for (int mt = 0; mt < 4; ++mt)
#pragma unroll
      for (int nt = 0; nt < 2; ++nt)
        acc[mt][nt] = wmma_bf16(af[mt], bf[nt], acc[mt][nt]);

    __syncthreads();
    cur ^= 1;
  }

  // epilogue: lane holds N = lane&15, rows M = (lane>>4)*8 + i
#pragma unroll
  for (int mt = 0; mt < 4; ++mt)
#pragma unroll
    for (int nt = 0; nt < 2; ++nt) {
      const size_t col = nBlk + waveN * 32 + nt * 16 + l15;
#pragma unroll
      for (int i = 0; i < 8; ++i) {
        const size_t row = mBlk + waveM * 64 + mt * 16 + half * 8 + i;
        const float v = acc[mt][nt][i];
        if (mode == 0) {
          Cb[row * D_MODEL + col] = f2bf(v);
        } else {
          Cf[row * D_MODEL + col] = v + bias[col] + resid[row * D_MODEL + col];
        }
      }
    }
}

// ------------------------------------------------------------------
// 4) Flash attention with relative-position skew term.
//    grid (T/128, H, B); 256 threads = 8 waves; wave w owns 16 query rows.
//    q/k/v/o layout: [B, T, H, Dh] bf16.
//    V staged transposed (Vt[n][kk], stride 40) so B-fragments are two
//    contiguous 16B ds_load_b128 per tile instead of 16 scalar loads.
// ------------------------------------------------------------------
struct __align__(16) AttnSmem {
  u16   Kb[32 * 64];       // K chunk, row-major [s][k]
  u16   Vt[64 * 40];       // V chunk transposed [n][kk], padded stride
  float scr[8 * 16 * 48];  // per-wave scratch: QE band, then P (bf16 view)
};

__global__ __launch_bounds__(256) void attn_kernel(const u16* __restrict__ q,
                                                   const u16* __restrict__ k,
                                                   const u16* __restrict__ v,
                                                   const u16* __restrict__ E,
                                                   u16* __restrict__ o) {
  __shared__ AttnSmem sm;
  const int tid  = threadIdx.x;
  const int lane = tid & 31;
  const int wave = tid >> 5;
  const int half = lane >> 4;
  const int l15  = lane & 15;

  const int qblk = blockIdx.x;
  const int h    = blockIdx.y;
  const int b    = blockIdx.z;
  const int t0   = qblk * 128;
  const int t0w  = t0 + wave * 16;

  float* ws_f = &sm.scr[wave * 16 * 48];
  u16*   ws_u = (u16*)ws_f;

  // Q fragments (16 rows x 64), A-operand layout
  u16x16 qa0, qa1;
  {
    const size_t qb = (((size_t)(b * TLEN + t0w + l15) * N_HEADS + h) * D_HEAD);
    qa0 = cat8(*(const u16x8*)&q[qb + half * 8],
               *(const u16x8*)&q[qb + 16 + half * 8]);
    qa1 = cat8(*(const u16x8*)&q[qb + 32 + half * 8],
               *(const u16x8*)&q[qb + 48 + half * 8]);
  }

  v8f   o_acc[4] = {};
  float m_i[8], l_i[8];
#pragma unroll
  for (int i = 0; i < 8; ++i) { m_i[i] = -1e30f; l_i[i] = 0.f; }

  const int sEnd = t0 + 128;
  for (int s0 = 0; s0 < sEnd; s0 += 32) {
    __syncthreads();
    // stage K chunk [32 x 64] row-major, V chunk transposed [64 x 32]
    {
      const int idx = tid * 8;
      const int r = idx >> 6, c = idx & 63;
      const size_t gb = (((size_t)(b * TLEN + s0 + r) * N_HEADS + h) * D_HEAD + c);
      *(u16x8*)&sm.Kb[r * 64 + c] = *(const u16x8*)&k[gb];
      u16x8 vv = *(const u16x8*)&v[gb];
#pragma unroll
      for (int j = 0; j < 8; ++j)
        sm.Vt[(c + j) * 40 + r] = vv[j];
      if (s0 + 32 < sEnd) {
        const size_t gn = gb + (size_t)32 * N_HEADS * D_HEAD;
        __builtin_prefetch(&k[gn], 0, 3);
        __builtin_prefetch(&v[gn], 0, 3);
      }
    }

    // QE band: QE[16,48] = Q @ E[r0 .. r0+47]^T ; rel[t,s] = QE[tc, 15+sc-tc]
    const int r0 = (TLEN - 1) + s0 - t0w - 15;
#pragma unroll
    for (int jt = 0; jt < 3; ++jt) {
      int rr = r0 + jt * 16 + l15;
      rr = rr < 0 ? 0 : (rr > 2 * TLEN - 3 ? 2 * TLEN - 3 : rr);
      const size_t eb = (size_t)rr * D_HEAD;
      u16x16 e0 = cat8(*(const u16x8*)&E[eb + half * 8],
                       *(const u16x8*)&E[eb + 16 + half * 8]);
      u16x16 e1 = cat8(*(const u16x8*)&E[eb + 32 + half * 8],
                       *(const u16x8*)&E[eb + 48 + half * 8]);
      v8f qe = {};
      qe = wmma_bf16(qa0, e0, qe);
      qe = wmma_bf16(qa1, e1, qe);
#pragma unroll
      for (int i = 0; i < 8; ++i)
        ws_f[(half * 8 + i) * 48 + jt * 16 + l15] = qe[i];
    }
    __syncthreads();   // K/V and QE band visible

    // S = Q K^T : [16 x 32]
    v8f sacc[2] = {};
#pragma unroll
    for (int nt = 0; nt < 2; ++nt) {
      const int sr = nt * 16 + l15;
      u16x16 kb0 = cat8(*(const u16x8*)&sm.Kb[sr * 64 + half * 8],
                        *(const u16x8*)&sm.Kb[sr * 64 + 16 + half * 8]);
      u16x16 kb1 = cat8(*(const u16x8*)&sm.Kb[sr * 64 + 32 + half * 8],
                        *(const u16x8*)&sm.Kb[sr * 64 + 48 + half * 8]);
      sacc[nt] = wmma_bf16(qa0, kb0, sacc[nt]);
      sacc[nt] = wmma_bf16(qa1, kb1, sacc[nt]);
    }

    // add rel, scale, causal mask, online softmax
    float sv[2][8], rm[8], alpha[8], pw[2][8], rs[8];
#pragma unroll
    for (int i = 0; i < 8; ++i) {
      const int tc = half * 8 + i;
      const int tg = t0w + tc;
#pragma unroll
      for (int nt = 0; nt < 2; ++nt) {
        const int sc = nt * 16 + l15;
        const int j  = 15 + sc - tc;                  // 0..46
        float val = (sacc[nt][i] + ws_f[tc * 48 + j]) * 0.125f;
        if (s0 + sc > tg) val = NEGINF;
        sv[nt][i] = val;
      }
      rm[i] = fmaxf(sv[0][i], sv[1][i]);
    }
#pragma unroll
    for (int msk = 1; msk < 16; msk <<= 1)
#pragma unroll
      for (int i = 0; i < 8; ++i)
        rm[i] = fmaxf(rm[i], __shfl_xor(rm[i], msk, 32));

#pragma unroll
    for (int i = 0; i < 8; ++i) {
      const float mn = fmaxf(m_i[i], rm[i]);
      alpha[i] = __expf(m_i[i] - mn);
      pw[0][i] = __expf(sv[0][i] - mn);
      pw[1][i] = __expf(sv[1][i] - mn);
      rs[i]    = pw[0][i] + pw[1][i];
      m_i[i]   = mn;
    }
#pragma unroll
    for (int msk = 1; msk < 16; msk <<= 1)
#pragma unroll
      for (int i = 0; i < 8; ++i)
        rs[i] += __shfl_xor(rs[i], msk, 32);
#pragma unroll
    for (int i = 0; i < 8; ++i) l_i[i] = l_i[i] * alpha[i] + rs[i];
#pragma unroll
    for (int nt = 0; nt < 4; ++nt)
#pragma unroll
      for (int i = 0; i < 8; ++i) o_acc[nt][i] *= alpha[i];

    __builtin_amdgcn_wave_barrier();  // keep rel reads before P stores
    // store P as bf16 [16 x 32] into wave scratch
#pragma unroll
    for (int i = 0; i < 8; ++i) {
      const int tc = half * 8 + i;
#pragma unroll
      for (int nt = 0; nt < 2; ++nt)
        ws_u[tc * 32 + nt * 16 + l15] = f2bf(pw[nt][i]);
    }
    __syncthreads();   // P visible

    // O += P V : P A-frag + Vt B-frags (contiguous in LDS)
    u16x16 pa = cat8(*(const u16x8*)&ws_u[l15 * 32 + half * 8],
                     *(const u16x8*)&ws_u[l15 * 32 + 16 + half * 8]);
#pragma unroll
    for (int nt = 0; nt < 4; ++nt) {
      const int n = nt * 16 + l15;
      u16x16 vb = cat8(*(const u16x8*)&sm.Vt[n * 40 + half * 8],
                       *(const u16x8*)&sm.Vt[n * 40 + 16 + half * 8]);
      o_acc[nt] = wmma_bf16(pa, vb, o_acc[nt]);
    }
  }

  // normalize and write out [B,T,H,Dh] bf16
  float inv[8];
#pragma unroll
  for (int i = 0; i < 8; ++i) inv[i] = 1.0f / l_i[i];
#pragma unroll
  for (int nt = 0; nt < 4; ++nt)
#pragma unroll
    for (int i = 0; i < 8; ++i) {
      const int row = t0w + half * 8 + i;
      const int col = nt * 16 + l15;
      o[(((size_t)(b * TLEN + row) * N_HEADS + h) * D_HEAD + col)] =
          f2bf(o_acc[nt][i] * inv[i]);
    }
}

// ------------------------------------------------------------------
// launch
// ------------------------------------------------------------------
extern "C" void kernel_launch(void* const* d_in, const int* in_sizes, int n_in,
                              void* d_out, int out_size, void* d_ws, size_t ws_size,
                              hipStream_t stream) {
  (void)in_sizes; (void)n_in; (void)out_size; (void)ws_size;
  const float* x     = (const float*)d_in[0];
  const float* ctx   = (const float*)d_in[1];
  const float* lut   = (const float*)d_in[2];
  // d_in[3] = mask (causal, derived analytically) -- unused
  const float* Wq    = (const float*)d_in[4];
  const float* Wk    = (const float*)d_in[5];
  const float* Wv    = (const float*)d_in[6];
  const float* Wo    = (const float*)d_in[7];
  const float* bo    = (const float*)d_in[8];
  const float* gamma = (const float*)d_in[9];
  const float* beta  = (const float*)d_in[10];

  char* ws = (char*)d_ws;
  size_t off = 0;
  const size_t MT = (size_t)BATCH * TLEN;               // 4096
  float* xn_f = (float*)(ws + off); off += MT * D_MODEL * 4;
  u16* xn_b   = (u16*)(ws + off);   off += MT * D_MODEL * 2;
  u16* ctx_b  = (u16*)(ws + off);   off += MT * D_MODEL * 2;
  u16* wq_b   = (u16*)(ws + off);   off += (size_t)D_MODEL * D_MODEL * 2;
  u16* wk_b   = (u16*)(ws + off);   off += (size_t)D_MODEL * D_MODEL * 2;
  u16* wv_b   = (u16*)(ws + off);   off += (size_t)D_MODEL * D_MODEL * 2;
  u16* wo_b   = (u16*)(ws + off);   off += (size_t)D_MODEL * D_MODEL * 2;
  u16* e_b    = (u16*)(ws + off);   off += 262144;
  u16* q_b    = (u16*)(ws + off);   off += MT * D_MODEL * 2;
  u16* k_b    = (u16*)(ws + off);   off += MT * D_MODEL * 2;
  u16* v_b    = (u16*)(ws + off);   off += MT * D_MODEL * 2;
  u16* at_b   = (u16*)(ws + off);   off += MT * D_MODEL * 2;

  ln_kernel<<<MT, 256, 0, stream>>>(x, gamma, beta, xn_f, xn_b);

  const int nCtx = BATCH * TLEN * D_MODEL;              // 4194304
  const int nW   = D_MODEL * D_MODEL;                   // 1048576
  const int nE   = (2 * TLEN - 1) * D_HEAD;             // 131008
  cvt_kernel<<<(nCtx + 255) / 256, 256, 0, stream>>>(ctx, ctx_b, nCtx);
  cvt_kernel<<<(nW + 255) / 256, 256, 0, stream>>>(Wq, wq_b, nW);
  cvt_kernel<<<(nW + 255) / 256, 256, 0, stream>>>(Wk, wk_b, nW);
  cvt_kernel<<<(nW + 255) / 256, 256, 0, stream>>>(Wv, wv_b, nW);
  cvt_kernel<<<(nW + 255) / 256, 256, 0, stream>>>(Wo, wo_b, nW);
  cvt_kernel<<<(nE + 255) / 256, 256, 0, stream>>>(lut, e_b, nE);

  dim3 ggrid(D_MODEL / 128, MT / 128);                  // (8, 32)
  gemm_bf16<<<ggrid, 256, 0, stream>>>(xn_b,  wq_b, q_b, nullptr, nullptr, nullptr, 0);
  gemm_bf16<<<ggrid, 256, 0, stream>>>(ctx_b, wk_b, k_b, nullptr, nullptr, nullptr, 0);
  gemm_bf16<<<ggrid, 256, 0, stream>>>(ctx_b, wv_b, v_b, nullptr, nullptr, nullptr, 0);

  attn_kernel<<<dim3(TLEN / 128, N_HEADS, BATCH), 256, 0, stream>>>(
      q_b, k_b, v_b, e_b, at_b);

  gemm_bf16<<<ggrid, 256, 0, stream>>>(at_b, wo_b, nullptr, (float*)d_out, bo, xn_f, 1);
}
